// KernelPredicting_83777632076429
// MI455X (gfx1250) — matrix operand order repaired and checked by reading, not compiled, not run on hardware
//
#include <hip/hip_runtime.h>
#include <hip/hip_bf16.h>

typedef __attribute__((ext_vector_type(16))) __bf16 v16bf;
typedef __attribute__((ext_vector_type(8)))  __bf16 v8bf;
typedef __attribute__((ext_vector_type(8)))  float  v8f;

#define NN     4
#define CC     16
#define HH     256
#define WW     256
#define KK     5
#define KK2    25          // 5*5 window positions
#define KDIM   144         // CC * 3 * 3 reduction dim
#define KSTEPS 5           // ceil(144/32) -> 160 padded
#define KPAD   168         // padded LDS row stride (bank-conflict free)
#define TILEW  64          // pixels per workgroup (4 waves x 16)
#define XCOLS  (TILEW + 4) // halo for 5x5 window

// ---------------------------------------------------------------------------
// Prep: reorder conv_w (400,16,3,3) fp32 into bf16 WMMA B-fragment blocks.
// Output channel permuted o' = u*16 + c so each 16-wide N-tile == one window
// position u across all 16 channels. Block(u,ks) = 512 bf16, lane-major:
// lane l holds column c=l&15, K = 32*ks + (t<8 ? 8*hi + t : 16 + 8*hi + t-8), hi=l>=16, t=0..15.
// Bias reordered to wsB[u*16+c] = conv_b[c*25+u].
// ---------------------------------------------------------------------------
__global__ __launch_bounds__(256)
void kpn_prep(const float* __restrict__ conv_w, const float* __restrict__ conv_b,
              __bf16* __restrict__ wsW, float* __restrict__ wsB) {
  int idx = blockIdx.x * blockDim.x + threadIdx.x;
  if (idx < KK2 * KSTEPS * 512) {
    int u  = idx / (KSTEPS * 512);
    int r  = idx % (KSTEPS * 512);
    int ks = r / 512;
    int e  = r % 512;
    int l  = e >> 4;
    int t  = e & 15;
    int c  = l & 15;
    int hi = l >> 4;
    int k  = 32 * ks + (t < 8 ? hi * 8 + t : 16 + hi * 8 + (t - 8));
    float val = 0.f;
    if (k < KDIM) {
      int ci = k / 9, tap = k % 9;
      int o  = c * KK2 + u;                      // original out-channel
      val = conv_w[((o * CC + ci) * 3 + tap / 3) * 3 + (tap % 3)];
    }
    wsW[idx] = (__bf16)val;
  }
  if (idx < CC * KK2) {
    int u = idx >> 4, c = idx & 15;
    wsB[idx] = conv_b[c * KK2 + u];
  }
}

// ---------------------------------------------------------------------------
// Fused: 3x3 conv (bf16 WMMA GEMM) -> online softmax over 25 window positions
// -> weighted 5x5 gather, all in registers. One workgroup = 64 pixels of one
// image row; each wave owns a 16-pixel M-tile and iterates all 25 N-tiles.
// ---------------------------------------------------------------------------
__global__ __launch_bounds__(128)
void kpn_main(const float* __restrict__ x,
              const __bf16* __restrict__ wsW, const float* __restrict__ wsB,
              float* __restrict__ out) {
  __shared__ float  xs[CC * KK * XCOLS];   // [c][dr 0..4][col] fp32 halo tile
  __shared__ __bf16 As[TILEW * KPAD];      // im2col panel [pixel][k]

  const int bid  = blockIdx.x;
  const int wseg = bid & 3;                // WW/TILEW = 4
  const int h    = (bid >> 2) & (HH - 1);
  const int n    = bid >> 10;
  const int w0   = wseg * TILEW;
  const int tid  = threadIdx.x;

  // Stage x halo tile (zero-padded borders)
  for (int i = tid; i < CC * KK * XCOLS; i += 128) {
    int c   = i / (KK * XCOLS);
    int rem = i % (KK * XCOLS);
    int dr  = rem / XCOLS;
    int col = rem % XCOLS;
    int gh  = h + dr - 2;
    int gw  = w0 + col - 2;
    float v = 0.f;
    if ((unsigned)gh < HH && (unsigned)gw < WW)
      v = x[((n * CC + c) * HH + gh) * WW + gw];
    xs[i] = v;
  }
  __syncthreads();

  // Build bf16 im2col panel from the fp32 tile (k = ci*9 + kh*3 + kw)
  for (int i = tid; i < TILEW * KPAD; i += 128) {
    int p = i / KPAD;
    int k = i % KPAD;
    float v = 0.f;
    if (k < KDIM) {
      int ci = k / 9, tap = k % 9;
      v = xs[(ci * KK + (tap / 3 + 1)) * XCOLS + (p + (tap % 3) + 1)];
    }
    As[i] = (__bf16)v;
  }
  __syncthreads();

  const int lane = tid & 31;
  const int wave = tid >> 5;       // wave32
  const int Mrow = lane & 15;      // A: row M, C/D: column N (= channel)
  const int hi   = lane >> 4;
  const int pix  = wave * 16;      // this wave's pixel base in the tile

  // A fragments: kept live in VGPRs across the whole 25-iteration u loop.
  // 16-bit A layout: lanes<16 hold K{0..7,16..23}, lanes>=16 hold K{8..15,24..31}.
  v16bf afrag[KSTEPS];
#pragma unroll
  for (int ks = 0; ks < KSTEPS; ks++) {
    const __bf16* base = &As[(pix + Mrow) * KPAD + 32 * ks];
    v8bf c0 = *(const v8bf*)(base + hi * 8);        // ds_load_b128
    v8bf c1 = *(const v8bf*)(base + 16 + hi * 8);   // ds_load_b128
#pragma unroll
    for (int t = 0; t < 8; t++) { afrag[ks][t] = c0[t]; afrag[ks][t + 8] = c1[t]; }
  }

  // Online-softmax state: per C-fragment row (8 pixels per lane)
  float m[8], ss[8], num[8];
#pragma unroll
  for (int r = 0; r < 8; r++) { m[r] = -3.0e38f; ss[r] = 0.f; num[r] = 0.f; }

  const int c = Mrow;  // channel owned by this lane in the C/D fragment

  for (int u = 0; u < KK2; u++) {
    // GEMM: logits for window position u, all 16 channels, 16 pixels
    v8f acc = {};
#pragma unroll
    for (int ks = 0; ks < KSTEPS; ks++) {
      const __bf16* bp = wsW + ((u * KSTEPS + ks) * 512 + lane * 16);
      v8bf b0 = *(const v8bf*)bp;          // global_load_b128 (L0/L2 resident)
      v8bf b1 = *(const v8bf*)(bp + 8);
      v16bf bfrag;
#pragma unroll
      for (int t = 0; t < 8; t++) { bfrag[t] = b0[t]; bfrag[t + 8] = b1[t]; }
      acc = __builtin_amdgcn_wmma_f32_16x16x32_bf16(
          false, afrag[ks], false, bfrag, (short)0, acc, false, false);
    }

    float bias = wsB[u * 16 + c];
    int du = u / KK, dv = u % KK;
    const float* xrow = &xs[(c * KK + du) * XCOLS + (pix + dv)];

#pragma unroll
    for (int r = 0; r < 8; r++) {
      int   p     = hi * 8 + r;                 // pixel index within M-tile
      float logit = acc[r] + bias;
      float v     = xrow[p];                    // unfold patch value
      float nm    = fmaxf(m[r], logit);
      float sc    = exp2f((m[r] - nm) * 1.44269504f);
      float e     = exp2f((logit - nm) * 1.44269504f);
      ss[r]  = ss[r]  * sc + e;
      num[r] = num[r] * sc + e * v;
      m[r]   = nm;
    }
  }

  // out[n][c][h][w] = weighted sum / softmax denominator
#pragma unroll
  for (int r = 0; r < 8; r++) {
    int p = pix + hi * 8 + r;
    out[((n * CC + c) * HH + h) * WW + (w0 + p)] = num[r] / ss[r];
  }
}

extern "C" void kernel_launch(void* const* d_in, const int* in_sizes, int n_in,
                              void* d_out, int out_size, void* d_ws, size_t ws_size,
                              hipStream_t stream) {
  const float* x      = (const float*)d_in[0];
  const float* conv_w = (const float*)d_in[1];
  const float* conv_b = (const float*)d_in[2];

  __bf16* wsW = (__bf16*)d_ws;                                   // 128000 B
  float*  wsB = (float*)((char*)d_ws + (size_t)KK2 * KSTEPS * 512 * sizeof(__bf16));

  kpn_prep<<<(KK2 * KSTEPS * 512 + 255) / 256, 256, 0, stream>>>(conv_w, conv_b, wsW, wsB);
  kpn_main<<<NN * HH * (WW / TILEW), 128, 0, stream>>>(x, wsW, wsB, (float*)d_out);
}
